// Model_17085379903565
// MI455X (gfx1250) — matrix operand order repaired and checked by reading, not compiled, not run on hardware
//
#include <hip/hip_runtime.h>
#include <hip/hip_bf16.h>
#include <math.h>

// ---------------------------------------------------------------------------
// N-BEATS/FNO spectral stack == linear frequency-diagonal operator.
// Active modes are confined to k in [0,128). Pipeline:
//   (0) x f32 -> f16
//   (1) build cos/sin basis tables (both orientations) in f16
//   (2) reduce w_real/w_imag over channels; scatter by idx to dense wmap
//   (3) WMMA analysis GEMM, split-K by 8:  Rp[s] partials
//       (M=1024, N=256, K=8192 -> 8 x K=1024)   [8192 waves]
//   (4) per-mode 9-block complex chain (sums the 8 partials) -> Y coeffs
//   (5) WMMA synthesis GEMM, B panel staged in LDS, fused epilogue writing
//       all 4 outputs with NT stores (M=2048, N=8192, K=256) [65536 waves]
// ---------------------------------------------------------------------------

typedef __attribute__((ext_vector_type(16))) _Float16 v16h;
typedef __attribute__((ext_vector_type(8)))  float    v8f;

#define B_SZ   1024
#define L_SZ   8192
#define NB     9
#define NM     64
#define NACT   128            // all active modes are < 128
#define NCOEF  256            // 128 modes * (re,im)
#define MSYN   2048           // backcast rows + forecast rows
#define SPLITK 8
#define KCHUNK (L_SZ / SPLITK)
#define SCALE_Y     268435456.0f      // 2^28: keep tiny coeffs in f16 range
#define INV_SCALE_Y (1.0f/268435456.0f)

union FragU { v16h h; unsigned u[8]; };

// A fragment (16x32 f16, M=lane%16): VGPR v -> K=2v,2v+1 (+16 for hi half-
// wave).  rowptr = start of this lane's row.
__device__ __forceinline__ v16h load_frag_a(const _Float16* rowptr, int k0, int lane) {
  const _Float16* p = rowptr + k0 + ((lane & 16) ? 16 : 0);
  FragU f;
  const unsigned* q0 = (const unsigned*)p;
  const unsigned* q1 = (const unsigned*)(p + 8);
#pragma unroll
  for (int v = 0; v < 4; ++v) { f.u[v] = q0[v]; f.u[4 + v] = q1[v]; }
  return f.h;
}

// B fragment (32x16 f16, N=lane%16): VGPR v -> K=2v,2v+1 (+16 for hi half-
// wave). nlineptr = start of this lane's N-line with K contiguous.
// (Works for both global and LDS source; addrspace inferred after inlining.)
template <typename P>
__device__ __forceinline__ v16h load_frag_b(const P* nlineptr, int k0, int lane) {
  const unsigned* q = (const unsigned*)(nlineptr + k0 + ((lane & 16) ? 16 : 0));
  FragU f;
#pragma unroll
  for (int v = 0; v < 8; ++v) f.u[v] = q[v];
  return f.h;
}

// ---------------------------------------------------------------------------
__global__ void k_cvt_x(const float* __restrict__ x, _Float16* __restrict__ xh, int n) {
  int i = blockIdx.x * blockDim.x + threadIdx.x;
  if (i < n) xh[i] = (_Float16)x[i];
}

// basis_tj[t*256 + 2m] = cos(2*pi*m*t/L), [2m+1] = sin(...)   (synthesis B)
// basis_jt[j*8192 + t] = same, transposed                     (analysis  B)
__global__ void k_basis(_Float16* __restrict__ basis_tj, _Float16* __restrict__ basis_jt) {
  int id = blockIdx.x * blockDim.x + threadIdx.x;   // [0, L_SZ*NACT)
  int m = id & (NACT - 1);
  int t = id >> 7;
  int r = (m * t) & (L_SZ - 1);                     // exact integer phase
  float th = (float)r * (6.283185307179586f / (float)L_SZ);
  float s, c;
  __sincosf(th, &s, &c);
  basis_tj[t * NCOEF + 2 * m]         = (_Float16)c;
  basis_tj[t * NCOEF + 2 * m + 1]     = (_Float16)s;
  basis_jt[(2 * m) * L_SZ + t]        = (_Float16)c;
  basis_jt[(2 * m + 1) * L_SZ + t]    = (_Float16)s;
}

__global__ void k_wzero(float* __restrict__ wmap) {
  int i = blockIdx.x * blockDim.x + threadIdx.x;
  if (i < NB * 3 * NACT * 2) wmap[i] = 0.0f;
}

// Sum complex weights over the 256-channel dim; scatter to dense k<128 map.
__global__ void k_wscatter(const float* __restrict__ wr, const float* __restrict__ wi,
                           const int* __restrict__ idx, float* __restrict__ wmap) {
  int id = blockIdx.x * blockDim.x + threadIdx.x;   // [0, 27*64)
  if (id >= NB * 3 * NM) return;
  int m  = id % NM;
  int bj = id / NM;                                  // (block*3 + role)
  float sr = 0.0f, si = 0.0f;
  for (int c = 0; c < 256; ++c) {
    sr += wr[(bj * 256 + c) * NM + m];
    si += wi[(bj * 256 + c) * NM + m];
  }
  int k = idx[bj * NM + m];
  if (k < NACT) {                                    // k>=128 can never act
    wmap[(bj * NACT + k) * 2]     = sr;
    wmap[(bj * NACT + k) * 2 + 1] = si;
  }
}

// Analysis GEMM, split-K: Rp[s](1024x256) partial = Xh[:,ks] * basis slice.
// One wave per (tile, split); 8192 waves total for latency hiding.
__global__ void k_analysis(const _Float16* __restrict__ xh,
                           const _Float16* __restrict__ basis_jt,
                           float* __restrict__ Rp) {
  int lane = threadIdx.x & 31;
  int wid  = threadIdx.x >> 5;
  int w    = blockIdx.x * 4 + wid;                   // [0, 8192)
  int tile = w & 1023;
  int s    = w >> 10;                                // split index
  int m0 = (tile >> 4) * 16;
  int n0 = (tile & 15) * 16;
  const _Float16* arow  = xh       + (size_t)(m0 + (lane & 15)) * L_SZ;
  const _Float16* bline = basis_jt + (size_t)(n0 + (lane & 15)) * L_SZ;
  int kbeg = s * KCHUNK, kend = kbeg + KCHUNK;
  v8f acc = {};
  for (int k0 = kbeg; k0 < kend; k0 += 32) {
    if ((k0 & 63) == 0) {                            // once per 128B cacheline
      // locality 3 -> WGP-scope global_prefetch_b8 (fills all cache levels);
      // speculative prefetch past the buffer end is silently dropped.
      __builtin_prefetch(arow  + k0 + 256, 0, 3);
      __builtin_prefetch(bline + k0 + 256, 0, 3);
    }
    v16h a = load_frag_a(arow,  k0, lane);
    v16h b = load_frag_b(bline, k0, lane);
    acc = __builtin_amdgcn_wmma_f32_16x16x32_f16(false, a, false, b,
                                                 (short)0, acc, false, false);
  }
  float* R = Rp + (size_t)s * B_SZ * NCOEF;
  int col   = n0 + (lane & 15);
  int rbase = m0 + ((lane & 16) ? 8 : 0);
#pragma unroll
  for (int v = 0; v < 8; ++v) R[(size_t)(rbase + v) * NCOEF + col] = acc[v];
}

// Per-(batch,mode) complex chain through the 9 blocks. Sums split-K partials
// first. Reference drops imag of bin 0 at every irfft -> realify k==0.
__global__ void k_chain(const float* __restrict__ Rp, const float* __restrict__ wmap,
                        _Float16* __restrict__ Y) {
  int id = blockIdx.x * blockDim.x + threadIdx.x;    // B*NACT
  int k = id & (NACT - 1);
  int b = id >> 7;
  if (b >= B_SZ) return;
  float br = 0.0f, bs = 0.0f;
#pragma unroll
  for (int s = 0; s < SPLITK; ++s) {
    const float* R = Rp + (size_t)s * B_SZ * NCOEF + (size_t)b * NCOEF;
    br += R[2 * k];
    bs += R[2 * k + 1];
  }
  float bi = -bs;                                    // xf imag = -sum x sin
  float x0r = br, x0i = bi;
  float fr = 0.0f, fi = 0.0f;
#pragma unroll
  for (int blk = 0; blk < NB; ++blk) {
    const float* w0 = wmap + (((blk * 3 + 0) * NACT) + k) * 2;
    const float* w1 = wmap + (((blk * 3 + 1) * NACT) + k) * 2;
    const float* w2 = wmap + (((blk * 3 + 2) * NACT) + k) * 2;
    float x1r = w0[0] * br - w0[1] * bi;
    float x1i = w0[0] * bi + w0[1] * br;
    if (k == 0) x1i = 0.0f;
    float cr = w1[0] * x1r - w1[1] * x1i;
    float ci = w1[0] * x1i + w1[1] * x1r;
    if (k == 0) ci = 0.0f;
    float gr = w2[0] * x1r - w2[1] * x1i;
    float gi = w2[0] * x1i + w2[1] * x1r;
    if (k == 0) gi = 0.0f;
    br -= cr; bi -= ci;
    fr += gr; fi += gi;
  }
  float corr_r = x0r - br, corr_i = x0i - bi;        // total backcast corr
  float alpha = ((k == 0) ? (1.0f / (float)L_SZ) : (2.0f / (float)L_SZ)) * SCALE_Y;
  Y[(size_t)b * NCOEF + 2 * k]                = (_Float16)( alpha * corr_r);
  Y[(size_t)b * NCOEF + 2 * k + 1]            = (_Float16)(-alpha * corr_i);
  Y[(size_t)(B_SZ + b) * NCOEF + 2 * k]       = (_Float16)( alpha * fr);
  Y[(size_t)(B_SZ + b) * NCOEF + 2 * k + 1]   = (_Float16)(-alpha * fi);
}

// Synthesis GEMM + fused epilogue: G(2048x8192) = Y * basis_tj-view.
// All 4 waves of a block share one 16(time)x256(coeff) B panel -> stage it
// in LDS once.  Rows <1024: backcast = x - G; rows >=1024: forecast = G.
__global__ void k_synth(const _Float16* __restrict__ Y,
                        const _Float16* __restrict__ basis_tj,
                        const float* __restrict__ x, float* __restrict__ out) {
  __shared__ _Float16 Bs[16 * NCOEF];                // 8 KB
  int lane = threadIdx.x & 31;
  int wid  = threadIdx.x >> 5;
  int n0 = blockIdx.x * 16;                          // time tile (512)
  int m0 = (blockIdx.y * 4 + wid) * 16;              // row tile  (128)

  // Cooperative stage: 128 threads x 64 B = 8 KB.
  {
    int t  = threadIdx.x;
    int n  = t >> 3;                                 // [0,16)
    int j0 = (t & 7) * 32;                           // [0,256) in steps of 32
    const uint4* src = (const uint4*)(basis_tj + (size_t)(n0 + n) * NCOEF + j0);
    uint4* dst = (uint4*)(&Bs[n * NCOEF + j0]);
#pragma unroll
    for (int q = 0; q < 4; ++q) dst[q] = src[q];
  }
  __syncthreads();

  const _Float16* arow  = Y + (size_t)(m0 + (lane & 15)) * NCOEF;
  const _Float16* bline = &Bs[(lane & 15) * NCOEF];
  v8f acc = {};
#pragma unroll
  for (int k0 = 0; k0 < NCOEF; k0 += 32) {
    v16h a = load_frag_a(arow,  k0, lane);
    v16h b = load_frag_b(bline, k0, lane);
    acc = __builtin_amdgcn_wmma_f32_16x16x32_f16(false, a, false, b,
                                                 (short)0, acc, false, false);
  }
  const size_t BL = (size_t)B_SZ * L_SZ;
  float* out_back = out;
  float* out_fore = out + BL;
  float* out_bm   = out + 2 * BL;
  float* out_fm   = out + 3 * BL;
  int col   = n0 + (lane & 15);
  int rbase = m0 + ((lane & 16) ? 8 : 0);
#pragma unroll
  for (int v = 0; v < 8; ++v) {
    int gm = rbase + v;
    float g = acc[v] * INV_SCALE_Y;
    if (gm < B_SZ) {                                 // uniform per tile
      size_t o = (size_t)gm * L_SZ + col;
      float bb = __builtin_nontemporal_load(&x[o]) - g;
      __builtin_nontemporal_store(bb, &out_back[o]);
      __builtin_nontemporal_store(bb * (1.0f / 9.0f), &out_bm[o]);
    } else {
      size_t o = (size_t)(gm - B_SZ) * L_SZ + col;
      __builtin_nontemporal_store(g, &out_fore[o]);
      __builtin_nontemporal_store(g * (1.0f / 9.0f), &out_fm[o]);
    }
  }
}

// ---------------------------------------------------------------------------
extern "C" void kernel_launch(void* const* d_in, const int* in_sizes, int n_in,
                              void* d_out, int out_size, void* d_ws, size_t ws_size,
                              hipStream_t stream) {
  (void)in_sizes; (void)n_in; (void)out_size; (void)ws_size;
  const float* x   = (const float*)d_in[0];
  const int*   idx = (const int*)d_in[1];
  const float* wr  = (const float*)d_in[2];
  const float* wi  = (const float*)d_in[3];
  float* out = (float*)d_out;

  char* ws = (char*)d_ws;
  _Float16* xh       = (_Float16*)ws; ws += (size_t)B_SZ * L_SZ * sizeof(_Float16);   // 16 MB
  _Float16* basis_tj = (_Float16*)ws; ws += (size_t)L_SZ * NCOEF * sizeof(_Float16);  //  4 MB
  _Float16* basis_jt = (_Float16*)ws; ws += (size_t)NCOEF * L_SZ * sizeof(_Float16);  //  4 MB
  float*    Rp       = (float*)ws;    ws += (size_t)SPLITK * B_SZ * NCOEF * sizeof(float); // 8 MB
  float*    wmap     = (float*)ws;    ws += (size_t)NB * 3 * NACT * 2 * sizeof(float);
  _Float16* Y        = (_Float16*)ws; ws += (size_t)MSYN * NCOEF * sizeof(_Float16);  //  1 MB

  k_cvt_x   <<<(B_SZ * L_SZ) / 256, 256, 0, stream>>>(x, xh, B_SZ * L_SZ);
  k_basis   <<<(L_SZ * NACT) / 256, 256, 0, stream>>>(basis_tj, basis_jt);
  k_wzero   <<<(NB * 3 * NACT * 2 + 255) / 256, 256, 0, stream>>>(wmap);
  k_wscatter<<<(NB * 3 * NM + 255) / 256, 256, 0, stream>>>(wr, wi, idx, wmap);
  k_analysis<<<(1024 * SPLITK) / 4, 128, 0, stream>>>(xh, basis_jt, Rp);
  k_chain   <<<(B_SZ * NACT) / 256, 256, 0, stream>>>(Rp, wmap, Y);
  dim3 sg(512, 32);
  k_synth   <<<sg, 128, 0, stream>>>(Y, basis_tj, x, out);
}
